// CorrelatedGraphConv_49357764166271
// MI455X (gfx1250) — matrix operand order repaired and checked by reading, not compiled, not run on hardware
//
#include <hip/hip_runtime.h>

// ---------------------------------------------------------------------------
// CDNA5 (gfx1250) implementation. All matmuls use v_wmma_f32_16x16x32_bf16
// (wave32). Per-wave tile: 64x32 = 4x2 WMMA fragments (1.5 b128 loads per
// WMMA); workgroup = 8 waves arranged 2(M) x 4(N) -> 128x128 tile.
// ---------------------------------------------------------------------------

typedef __bf16 bf16_t;
typedef __attribute__((ext_vector_type(8)))  __bf16 v8bf;
typedef __attribute__((ext_vector_type(16))) __bf16 v16bf;
typedef __attribute__((ext_vector_type(8)))  float  v8f;

#define DIM   1024
#define NB    256
#define NOBJ  100
#define MROWS (NB * NOBJ)   // 25600 flattened feature rows
#define RPAD  112           // padded row count (n) for S / alpha
#define CPAD  128           // padded col count (m) -> K-dim of aggregation

// A-fragment (16x32 bf16, MxK): lane<16 holds M=lane, elems 0..7 = K[k0..k0+7],
// elems 8..15 = K[k0+16..k0+23]; lanes>=16 are the +8 K-offset halves.
static __device__ __forceinline__ v16bf load_a_frag(const bf16_t* p) {
  v8bf lo = *(const v8bf*)(p);
  v8bf hi = *(const v8bf*)(p + 16);
  return __builtin_shufflevector(lo, hi, 0,1,2,3,4,5,6,7,8,9,10,11,12,13,14,15);
}
// B-fragment (32x16 bf16, KxN): lane<16 holds N=lane, elems 0..7 = K[k0..k0+7],
// elems 8..15 = K[k0+16..k0+23] (from a pre-transposed B, i.e. row n holds contiguous K).
static __device__ __forceinline__ v16bf load_b_frag(const bf16_t* p) {
  v8bf lo = *(const v8bf*)(p);
  v8bf hi = *(const v8bf*)(p + 16);
  return __builtin_shufflevector(lo, hi, 0,1,2,3,4,5,6,7,8,9,10,11,12,13,14,15);
}
static __device__ __forceinline__ v8f wmma_bf16(v16bf a, v16bf b, v8f c) {
  return __builtin_amdgcn_wmma_f32_16x16x32_bf16(false, a, false, b,
                                                 (short)0, c, false, false);
}
static __device__ __forceinline__ int imin(int a, int b) { return a < b ? a : b; }

// ---------------------------------------------------------------------------
// 1) feature f32 -> bf16 (vectorized)
__global__ void cast_feat_kernel(const float* __restrict__ in,
                                 bf16_t* __restrict__ out) {
  size_t i = ((size_t)blockIdx.x * blockDim.x + threadIdx.x) * 4;
  float4 v = *(const float4*)(in + i);
  out[i + 0] = (bf16_t)v.x;
  out[i + 1] = (bf16_t)v.y;
  out[i + 2] = (bf16_t)v.z;
  out[i + 3] = (bf16_t)v.w;
}

// 2) transpose+cast the three weight matrices: Wt[w][o][i] = W_w[i][o] (bf16)
__global__ void transpose_w_kernel(const float* __restrict__ w0,
                                   const float* __restrict__ w1,
                                   const float* __restrict__ w2,
                                   bf16_t* __restrict__ Wt) {
  size_t t = (size_t)blockIdx.x * blockDim.x + threadIdx.x;  // 3*DIM*DIM
  int w   = (int)(t / (DIM * DIM));
  int rem = (int)(t % (DIM * DIM));
  int i = rem / DIM;   // input row   (slow)
  int o = rem % DIM;   // input col   (fast -> coalesced read)
  const float* src = (w == 0) ? w0 : (w == 1) ? w1 : w2;
  Wt[(size_t)w * DIM * DIM + (size_t)o * DIM + i] = (bf16_t)src[(size_t)i * DIM + o];
}

// 3) zero the padded m-columns (100..127) of Yt so alpha(=0)*pad stays finite
__global__ void zero_yt_pad_kernel(bf16_t* __restrict__ Yt) {
  size_t t = (size_t)blockIdx.x * blockDim.x + threadIdx.x;  // NB*DIM*28
  size_t bo = t / 28;
  int m = NOBJ + (int)(t % 28);
  Yt[bo * CPAD + m] = (bf16_t)0.0f;
}

// ---------------------------------------------------------------------------
// 4) Projections: [25600,1024] x [1024,1024] for w in {weight, wq, wk}.
//    w==0 -> scatter transposed into Yt[b][o][m];  w==1 -> Q;  w==2 -> K.
//    Grid: (MROWS/128, DIM/128, 3); per-wave 64x32, waves 2(M) x 4(N).
__global__ void __launch_bounds__(256) proj_gemm_kernel(
    const bf16_t* __restrict__ A,    // feat bf16 [MROWS][DIM]
    const bf16_t* __restrict__ Wt,   // [3][DIM][DIM] (pre-transposed)
    bf16_t* __restrict__ Q,          // [MROWS][DIM]
    bf16_t* __restrict__ Kmat,       // [MROWS][DIM]
    bf16_t* __restrict__ Yt)         // [NB][DIM][CPAD]
{
  const int w    = blockIdx.z;
  const bf16_t* Bt = Wt + (size_t)w * DIM * DIM;
  const int wave = threadIdx.x >> 5;
  const int lane = threadIdx.x & 31;
  const int wm = wave >> 2, wn = wave & 3;
  const int l16 = lane & 15, half = lane >> 4;
  const int m0 = blockIdx.x * 128 + wm * 64;
  const int n0 = blockIdx.y * 128 + wn * 32;

  const bf16_t* arow[4];
  const bf16_t* brow[2];
#pragma unroll
  for (int f = 0; f < 4; ++f)
    arow[f] = A + (size_t)(m0 + f * 16 + l16) * DIM + half * 8;
#pragma unroll
  for (int g = 0; g < 2; ++g)
    brow[g] = Bt + (size_t)(n0 + g * 16 + l16) * DIM + half * 8;

  v8f acc[4][2] = {};
  for (int k = 0; k < DIM; k += 32) {
    v16bf af[4], bf[2];
#pragma unroll
    for (int f = 0; f < 4; ++f) af[f] = load_a_frag(arow[f] + k);
#pragma unroll
    for (int g = 0; g < 2; ++g) bf[g] = load_b_frag(brow[g] + k);
#pragma unroll
    for (int f = 0; f < 4; ++f)
#pragma unroll
      for (int g = 0; g < 2; ++g)
        acc[f][g] = wmma_bf16(af[f], bf[g], acc[f][g]);
  }

  // C/D layout: VGPR r, lane<16 -> (M=r, N=lane); lane>=16 -> (M=r+8, N=lane-16)
#pragma unroll
  for (int f = 0; f < 4; ++f) {
#pragma unroll
    for (int g = 0; g < 2; ++g) {
#pragma unroll
      for (int r = 0; r < 8; ++r) {
        int row = m0 + f * 16 + r + half * 8;   // global feature row
        int col = n0 + g * 16 + l16;            // output dim o
        float v = acc[f][g][r];
        if (w == 0) {
          unsigned gg = (unsigned)row;
          unsigned b = gg / 100u, mm = gg % 100u;
          Yt[((size_t)b * DIM + col) * CPAD + mm] = (bf16_t)v;
        } else if (w == 1) {
          Q[(size_t)row * DIM + col] = (bf16_t)v;
        } else {
          Kmat[(size_t)row * DIM + col] = (bf16_t)v;
        }
      }
    }
  }
}

// ---------------------------------------------------------------------------
// 5) S = Q * K^T per batch: one WG covers the whole 128x128 (padded) tile.
//    Grid: (NB); per-wave 64x32, waves 2(M) x 4(N).
__global__ void __launch_bounds__(256) s_gemm_kernel(
    const bf16_t* __restrict__ Q, const bf16_t* __restrict__ Kmat,
    float* __restrict__ S)
{
  const int b    = blockIdx.x;
  const int wave = threadIdx.x >> 5;
  const int lane = threadIdx.x & 31;
  const int wm = wave >> 2, wn = wave & 3;
  const int l16 = lane & 15, half = lane >> 4;
  const int n0 = wm * 64;   // S rows (query index n)
  const int m0 = wn * 32;   // S cols (key index m)

  const size_t base = (size_t)b * NOBJ;
  const bf16_t* arow[4];
  const bf16_t* brow[2];
#pragma unroll
  for (int f = 0; f < 4; ++f)
    arow[f] = Q + (base + imin(n0 + f * 16 + l16, NOBJ - 1)) * DIM + half * 8;
#pragma unroll
  for (int g = 0; g < 2; ++g)
    brow[g] = Kmat + (base + imin(m0 + g * 16 + l16, NOBJ - 1)) * DIM + half * 8;

  v8f acc[4][2] = {};
  for (int k = 0; k < DIM; k += 32) {
    v16bf af[4], bf[2];
#pragma unroll
    for (int f = 0; f < 4; ++f) af[f] = load_a_frag(arow[f] + k);
#pragma unroll
    for (int g = 0; g < 2; ++g) bf[g] = load_b_frag(brow[g] + k);
#pragma unroll
    for (int f = 0; f < 4; ++f)
#pragma unroll
      for (int g = 0; g < 2; ++g)
        acc[f][g] = wmma_bf16(af[f], bf[g], acc[f][g]);
  }

#pragma unroll
  for (int f = 0; f < 4; ++f)
#pragma unroll
    for (int g = 0; g < 2; ++g)
#pragma unroll
      for (int r = 0; r < 8; ++r) {
        int n = n0 + f * 16 + r + half * 8;
        int m = m0 + g * 16 + l16;
        if (n < RPAD) S[((size_t)b * RPAD + n) * CPAD + m] = acc[f][g][r];
      }
}

// ---------------------------------------------------------------------------
// 6) relu + adjacency mask + softmax over axis n (dim=1), per column m.
//    Writes alpha[b][112][128] bf16 fully (pads zeroed).
__global__ void softmax_kernel(const float* __restrict__ S,
                               const int* __restrict__ graph,
                               bf16_t* __restrict__ alpha)
{
  int b = blockIdx.x, m = threadIdx.x;  // m in 0..127
  const float* s = S     + (size_t)b * RPAD * CPAD + m;
  bf16_t*      a = alpha + (size_t)b * RPAD * CPAD + m;
  if (m < NOBJ) {
    const int* g = graph + (size_t)b * NOBJ * NOBJ + m;
    float mx = -3.0e38f, sum = 0.0f;
    for (int n = 0; n < NOBJ; ++n) {
      float v = s[(size_t)n * CPAD];
      v = (v > 0.0f && g[n * NOBJ] != 0) ? v : 0.0f;  // relu * mask
      if (v > mx) { sum *= __expf(mx - v); mx = v; }
      sum += __expf(v - mx);
    }
    float inv = 1.0f / sum;
    for (int n = 0; n < NOBJ; ++n) {
      float v = s[(size_t)n * CPAD];
      v = (v > 0.0f && g[n * NOBJ] != 0) ? v : 0.0f;
      a[(size_t)n * CPAD] = (bf16_t)(__expf(v - mx) * inv);
    }
    for (int n = NOBJ; n < RPAD; ++n) a[(size_t)n * CPAD] = (bf16_t)0.0f;
  } else {
    for (int n = 0; n < RPAD; ++n) a[(size_t)n * CPAD] = (bf16_t)0.0f;
  }
}

// ---------------------------------------------------------------------------
// 7) out[b][n][o] = sum_m alpha[b][n][m] * Y[b][m][o], K-dim = 128 (padded m).
//    B operand is Yt[b][o][m] (pre-transposed by the projection kernel).
//    Grid: (NB, DIM/128); per-wave 64x32, waves 2(M) x 4(N).
__global__ void __launch_bounds__(256) agg_gemm_kernel(
    const bf16_t* __restrict__ alpha, const bf16_t* __restrict__ Yt,
    float* __restrict__ out)
{
  const int b    = blockIdx.x;
  const int wave = threadIdx.x >> 5;
  const int lane = threadIdx.x & 31;
  const int wm = wave >> 2, wn = wave & 3;
  const int l16 = lane & 15, half = lane >> 4;
  const int n0 = wm * 64;                     // out rows (pad clamped)
  const int o0 = blockIdx.y * 128 + wn * 32;  // out cols

  const bf16_t* Ab = alpha + (size_t)b * RPAD * CPAD;
  const bf16_t* Bb = Yt    + (size_t)b * DIM * CPAD;
  const bf16_t* arow[4];
  const bf16_t* brow[2];
#pragma unroll
  for (int f = 0; f < 4; ++f)
    arow[f] = Ab + (size_t)imin(n0 + f * 16 + l16, RPAD - 1) * CPAD + half * 8;
#pragma unroll
  for (int g = 0; g < 2; ++g)
    brow[g] = Bb + (size_t)(o0 + g * 16 + l16) * CPAD + half * 8;

  v8f acc[4][2] = {};
#pragma unroll
  for (int k = 0; k < CPAD; k += 32) {
    v16bf af[4], bf[2];
#pragma unroll
    for (int f = 0; f < 4; ++f) af[f] = load_a_frag(arow[f] + k);
#pragma unroll
    for (int g = 0; g < 2; ++g) bf[g] = load_b_frag(brow[g] + k);
#pragma unroll
    for (int f = 0; f < 4; ++f)
#pragma unroll
      for (int g = 0; g < 2; ++g)
        acc[f][g] = wmma_bf16(af[f], bf[g], acc[f][g]);
  }

#pragma unroll
  for (int f = 0; f < 4; ++f)
#pragma unroll
    for (int g = 0; g < 2; ++g)
#pragma unroll
      for (int r = 0; r < 8; ++r) {
        int n = n0 + f * 16 + r + half * 8;
        int o = o0 + g * 16 + l16;
        if (n < NOBJ) out[((size_t)b * NOBJ + n) * DIM + o] = acc[f][g][r];
      }
}

// ---------------------------------------------------------------------------
extern "C" void kernel_launch(void* const* d_in, const int* in_sizes, int n_in,
                              void* d_out, int out_size, void* d_ws, size_t ws_size,
                              hipStream_t stream) {
  const float* feature = (const float*)d_in[0];
  const int*   graph   = (const int*)d_in[1];
  const float* weight  = (const float*)d_in[2];
  const float* wq      = (const float*)d_in[3];
  const float* wk      = (const float*)d_in[4];

  // workspace layout (~148 MB total)
  char* ws = (char*)d_ws;
  bf16_t* featbf = (bf16_t*)ws;  ws += (size_t)MROWS * DIM * sizeof(bf16_t);     // 52.4 MB
  bf16_t* Wt     = (bf16_t*)ws;  ws += (size_t)3 * DIM * DIM * sizeof(bf16_t);   //  6.3 MB
  bf16_t* Yt     = (bf16_t*)ws;  ws += (size_t)NB * DIM * CPAD * sizeof(bf16_t); // 67.1 MB
  float*  S      = (float*)ws;   ws += (size_t)NB * RPAD * CPAD * sizeof(float); // 14.7 MB
  bf16_t* alpha  = (bf16_t*)ws;  ws += (size_t)NB * RPAD * CPAD * sizeof(bf16_t);//  7.3 MB

  // Q/K bf16 scratch exactly fills d_out (dead before agg_gemm overwrites it)
  bf16_t* Q    = (bf16_t*)d_out;
  bf16_t* Kmat = Q + (size_t)MROWS * DIM;

  cast_feat_kernel  <<<MROWS * DIM / 1024, 256, 0, stream>>>(feature, featbf);
  transpose_w_kernel<<<3 * DIM * DIM / 256, 256, 0, stream>>>(weight, wq, wk, Wt);
  zero_yt_pad_kernel<<<NB * DIM * 28 / 256, 256, 0, stream>>>(Yt);
  proj_gemm_kernel  <<<dim3(MROWS / 128, DIM / 128, 3), 256, 0, stream>>>(featbf, Wt, Q, Kmat, Yt);
  s_gemm_kernel     <<<NB, 256, 0, stream>>>(Q, Kmat, S);
  softmax_kernel    <<<NB, CPAD, 0, stream>>>(S, graph, alpha);
  agg_gemm_kernel   <<<dim3(NB, DIM / 128), 256, 0, stream>>>(alpha, Yt, (float*)d_out);
}
